// PBiLSTM_CRF_88476326297869
// MI455X (gfx1250) — matrix operand order repaired and checked by reading, not compiled, not run on hardware
//
#include <hip/hip_runtime.h>
#include <stdint.h>

#define LBL 512
#define DIM 2048
#define SEQT 8192
#define NEGV (-10000.0f)

typedef __attribute__((ext_vector_type(16))) __bf16        v16bf;
typedef __attribute__((ext_vector_type(8)))  float         v8f;
typedef __attribute__((ext_vector_type(4)))  unsigned int  v4u;
typedef __attribute__((ext_vector_type(8)))  int           v8i;
typedef __attribute__((ext_vector_type(4)))  int           v4i;

// ---- f32 -> bf16 bits (round to nearest even) ----
__device__ __forceinline__ unsigned short bfbits(float f) {
    unsigned u = __builtin_bit_cast(unsigned, f);
    unsigned r = u + 0x7FFFu + ((u >> 16) & 1u);
    return (unsigned short)(r >> 16);
}
__device__ __forceinline__ unsigned bfpack2(float lo, float hi) {
    return (unsigned)bfbits(lo) | ((unsigned)bfbits(hi) << 16);
}

// ---- Tensor Data Mover: 2D tile global -> LDS ----
// ds_code: 0=1B, 1=2B, 2=4B elements. Tile fully in-bounds (tensor_dim == tile_dim).
__device__ __forceinline__ void tdm_load_tile_2d(uint32_t lds_addr, const void* gptr,
                                                 uint32_t tile_w, uint32_t tile_h,
                                                 uint32_t row_stride_elems,
                                                 uint32_t ds_code) {
    uint64_t ga = (uint64_t)(uintptr_t)gptr;
    v4u g0;
    g0[0] = 1u;                                                // count=1 user descriptor
    g0[1] = lds_addr;                                          // lds_addr[31:0]
    g0[2] = (uint32_t)ga;                                      // global_addr[31:0]
    g0[3] = (uint32_t)((ga >> 32) & 0x01FFFFFFu) | (2u << 30); // global_addr hi | type=2
    v8i g1;
    g1[0] = (int)(ds_code << 16);                              // data_size
    g1[1] = (int)((tile_w & 0xFFFFu) << 16);                   // tensor_dim0[15:0]
    g1[2] = (int)((tile_w >> 16) | ((tile_h & 0xFFFFu) << 16));// tensor_dim0 hi | tensor_dim1 lo
    g1[3] = (int)((tile_h >> 16) | ((tile_w & 0xFFFFu) << 16));// tensor_dim1 hi | tile_dim0
    g1[4] = (int)(tile_h & 0xFFFFu);                           // tile_dim1 | tile_dim2=0
    g1[5] = (int)row_stride_elems;                             // tensor_dim0_stride[31:0]
    g1[6] = 0;
    g1[7] = 0;
    v4i z4 = {0, 0, 0, 0};
    v8i z8 = {0, 0, 0, 0, 0, 0, 0, 0};
    __builtin_amdgcn_tensor_load_to_lds(g0, g1, z4, z4, z8, 0);
}

// =====================================================================
// Kernel 0a: x (f32) -> xbf (bf16), elementwise, 8 elems/thread
// =====================================================================
__global__ __launch_bounds__(256) void k_conv_x(const float* __restrict__ x,
                                                unsigned short* __restrict__ xbf) {
    const size_t i = ((size_t)blockIdx.x * 256 + threadIdx.x) * 8;
    const float4 f0 = *(const float4*)(x + i);
    const float4 f1 = *(const float4*)(x + i + 4);
    uint4 o;
    o.x = bfpack2(f0.x, f0.y);
    o.y = bfpack2(f0.z, f0.w);
    o.z = bfpack2(f1.x, f1.y);
    o.w = bfpack2(f1.z, f1.w);
    *(uint4*)(xbf + i) = o;
}

// =====================================================================
// Kernel 0b: Wt[n][k] = bf16(W[k][n])  (transpose + convert, coalesced writes)
// =====================================================================
__global__ __launch_bounds__(256) void k_conv_wt(const float* __restrict__ W,
                                                 unsigned short* __restrict__ Wt) {
    const int k = blockIdx.x * 256 + threadIdx.x;
    const int n = blockIdx.y;
    Wt[(size_t)n * DIM + k] = bfbits(W[(size_t)k * LBL + n]);
}

// =====================================================================
// Kernel 1: feats[T,L] = x @ W + b.  bf16 WMMA, TDM-fed double-buffered LDS.
// 256 threads = 8 waves; block tile 64x64; BK=64.
// Waves 0-3 DMA A slices (xbf rows), waves 4-7 DMA B slices (Wt rows);
// both operands have row stride DIM, so one branchless TDM call per wave.
// Each wave owns a 16(M) x 32(N) slab -> 2 accumulators, 4 WMMA per BK step.
// =====================================================================
__global__ __launch_bounds__(256) void k_gemm_feats(const unsigned short* __restrict__ xbf,
                                                    const unsigned short* __restrict__ Wt,
                                                    const float* __restrict__ bias,
                                                    float* __restrict__ feats) {
    __shared__ unsigned short sA[2][64 * 64];  // [m][k]
    __shared__ unsigned short sB[2][64 * 64];  // [n][k]

    const int tid  = threadIdx.x;
    const int wave = tid >> 5;
    const int lane = tid & 31;
    const int bm = blockIdx.x * 64;
    const int bn = blockIdx.y * 64;
    const int wm = (wave & 3) * 16;
    const int wn = (wave >> 2) * 32;
    const int hs = lane >> 4;        // 0: lanes 0-15, 1: lanes 16-31
    const int lm = lane & 15;

    // Branchless per-wave DMA slice parameters (A for waves 0-3, B for waves 4-7).
    const bool isA = wave < 4;
    const int  slice = isA ? wave : (wave - 4);
    const unsigned short* gsrc = (isA ? xbf + (uint64_t)(bm + slice * 16) * DIM
                                      : Wt  + (uint64_t)(bn + slice * 16) * DIM);
    unsigned short* ldsdst0 = isA ? &sA[0][slice * 16 * 64] : &sB[0][slice * 16 * 64];
    unsigned short* ldsdst1 = isA ? &sA[1][slice * 16 * 64] : &sB[1][slice * 16 * 64];

    v8f c0 = {};
    v8f c1 = {};

    union Frag { v16bf v; uint4 q[2]; };

    // preload tile 0 into buffer 0
    tdm_load_tile_2d((uint32_t)(uintptr_t)ldsdst0, gsrc, 64u, 16u, (uint32_t)DIM, 1u);
    __builtin_amdgcn_s_wait_tensorcnt(0);
    __syncthreads();

    const int NIT = DIM / 64;
    for (int it = 0; it < NIT; ++it) {
        const int buf = it & 1;
        if (it + 1 < NIT) {
            unsigned short* nxt = (buf == 0) ? ldsdst1 : ldsdst0;
            tdm_load_tile_2d((uint32_t)(uintptr_t)nxt, gsrc + (it + 1) * 64,
                             64u, 16u, (uint32_t)DIM, 1u);
        }

        const unsigned short* tA = sA[buf];
        const unsigned short* tB = sB[buf];
#pragma unroll
        for (int ks = 0; ks < 64; ks += 32) {
            Frag fa, fb0, fb1;
            // A fragment: lane m = wm+lm; 16 contiguous K starting at ks + hs*16
            fa.q[0] = *(const uint4*)&tA[(wm + lm) * 64 + ks + hs * 16];
            fa.q[1] = *(const uint4*)&tA[(wm + lm) * 64 + ks + hs * 16 + 8];
            // B fragments: lane n; 16 contiguous K starting at ks + hs*16
            const int bo = ks + hs * 16;
            fb0.q[0] = *(const uint4*)&tB[(wn + lm) * 64 + bo];
            fb0.q[1] = *(const uint4*)&tB[(wn + lm) * 64 + bo + 8];
            fb1.q[0] = *(const uint4*)&tB[(wn + 16 + lm) * 64 + bo];
            fb1.q[1] = *(const uint4*)&tB[(wn + 16 + lm) * 64 + bo + 8];
            c0 = __builtin_amdgcn_wmma_f32_16x16x32_bf16(false, fa.v, false, fb0.v, (short)0, c0, false, false);
            c1 = __builtin_amdgcn_wmma_f32_16x16x32_bf16(false, fa.v, false, fb1.v, (short)0, c1, false, false);
        }

        __builtin_amdgcn_s_wait_tensorcnt(0);  // next buffer's DMA landed (this wave)
        __syncthreads();                       // all waves: DMA landed + LDS reads done
    }

    // epilogue: C layout — VGPR r, lane L: M = r + 8*hs, N = L%16
    const int gm = bm + wm + hs * 8;
    const int gn = bn + wn + lm;
    const float bz0 = bias[gn];
    const float bz1 = bias[gn + 16];
#pragma unroll
    for (int r = 0; r < 8; ++r) {
        feats[(uint64_t)(gm + r) * LBL + gn]      = c0[r] + bz0;
        feats[(uint64_t)(gm + r) * LBL + gn + 16] = c1[r] + bz1;
    }
}

// =====================================================================
// Kernel 2: mT[j] = max_i trans[j,i];  Et[i,j] = exp(trans[j,i] - mT[j])
// =====================================================================
__global__ __launch_bounds__(256) void k_prep(const float* __restrict__ trans,
                                              float* __restrict__ Et,
                                              float* __restrict__ mT) {
    __shared__ float red[256];
    const int j = blockIdx.x;
    const int tid = threadIdx.x;
    float m = -3.4e38f;
    for (int i = tid; i < LBL; i += 256) m = fmaxf(m, trans[j * LBL + i]);
    red[tid] = m;
    __syncthreads();
    for (int s = 128; s > 0; s >>= 1) {
        if (tid < s) red[tid] = fmaxf(red[tid], red[tid + s]);
        __syncthreads();
    }
    m = red[0];
    if (tid == 0) mT[j] = m;
    for (int i = tid; i < LBL; i += 256)
        Et[(uint64_t)i * LBL + j] = __expf(trans[j * LBL + i] - m);
}

// =====================================================================
// Kernel 3: gold path score (single block, deterministic reduction)
// =====================================================================
__global__ __launch_bounds__(256) void k_gold(const float* __restrict__ feats,
                                              const float* __restrict__ trans,
                                              const int* __restrict__ tags,
                                              const int* __restrict__ start_id,
                                              const int* __restrict__ stop_id,
                                              float* __restrict__ gold) {
    __shared__ float red[256];
    const int tid = threadIdx.x;
    const int start = start_id[0];
    const int stop  = stop_id[0];
    float s = 0.f;
    for (int t = tid; t < SEQT; t += 256) {
        const int cur  = tags[t];
        const int prev = (t == 0) ? start : tags[t - 1];
        s += feats[(uint64_t)t * LBL + cur] + trans[cur * LBL + prev];
    }
    if (tid == 0) s += trans[stop * LBL + tags[SEQT - 1]];
    red[tid] = s;
    __syncthreads();
    for (int st = 128; st > 0; st >>= 1) {
        if (tid < st) red[tid] += red[tid + st];
        __syncthreads();
    }
    if (tid == 0) gold[0] = red[0];
}

// =====================================================================
// Kernel 4: sequential CRF forward scan (one persistent 512-thread WG).
// alpha'[j] = feat[t,j] + m_a + mT[j] + log( sum_i Et[i,j]*exp(alpha[i]-m_a) )
// =====================================================================
__global__ __launch_bounds__(512) void k_scan(const float* __restrict__ feats,
                                              const float* __restrict__ Et,
                                              const float* __restrict__ mT,
                                              const float* __restrict__ trans,
                                              const int* __restrict__ start_id,
                                              const int* __restrict__ stop_id,
                                              const float* __restrict__ gold,
                                              float* __restrict__ out) {
    __shared__ float alpha[LBL];
    __shared__ float ea[LBL];
    __shared__ float red[LBL];
    __shared__ float sm;
    const int j = threadIdx.x;
    const int start = start_id[0];
    const int stop  = stop_id[0];

    alpha[j] = (j == start) ? 0.f : NEGV;
    if (j == 0) sm = 0.f;            // max(init_alpha) = 0
    const float mTj = mT[j];
    __syncthreads();

    for (int t = 0; t < SEQT; ++t) {
        if (t + 1 < SEQT)
            __builtin_prefetch(&feats[(uint64_t)(t + 1) * LBL + j], 0, 0);

        const float m_a = sm;
        ea[j] = __expf(alpha[j] - m_a);
        __syncthreads();

        float y = 0.f;
        const float* __restrict__ Ecol = Et + j;
#pragma unroll 8
        for (int i = 0; i < LBL; ++i)
            y = __fmaf_rn(Ecol[(uint64_t)i * LBL], ea[i], y);

        const float an = feats[(uint64_t)t * LBL + j] + m_a + mTj + __logf(y);
        alpha[j] = an;
        red[j]   = an;
        __syncthreads();
        for (int s2 = 256; s2 > 0; s2 >>= 1) {
            if (j < s2) red[j] = fmaxf(red[j], red[j + s2]);
            __syncthreads();
        }
        if (j == 0) sm = red[0];
        __syncthreads();
    }

    // forward_score = logsumexp_j(alpha[j] + trans[stop, j])
    const float v = alpha[j] + trans[stop * LBL + j];
    red[j] = v;
    __syncthreads();
    for (int s2 = 256; s2 > 0; s2 >>= 1) {
        if (j < s2) red[j] = fmaxf(red[j], red[j + s2]);
        __syncthreads();
    }
    const float m = red[0];
    __syncthreads();
    red[j] = __expf(v - m);
    __syncthreads();
    for (int s2 = 256; s2 > 0; s2 >>= 1) {
        if (j < s2) red[j] += red[j + s2];
        __syncthreads();
    }
    if (j == 0) out[0] = (m + __logf(red[0])) - gold[0];
}

extern "C" void kernel_launch(void* const* d_in, const int* in_sizes, int n_in,
                              void* d_out, int out_size, void* d_ws, size_t ws_size,
                              hipStream_t stream) {
    (void)in_sizes; (void)n_in; (void)out_size; (void)ws_size;
    const float* x        = (const float*)d_in[0];
    const float* W        = (const float*)d_in[1];
    const float* bias     = (const float*)d_in[2];
    const float* trans    = (const float*)d_in[3];
    const int*   tags     = (const int*)d_in[4];
    const int*   start_id = (const int*)d_in[5];
    const int*   stop_id  = (const int*)d_in[6];

    float* wsf = (float*)d_ws;
    size_t o = 0;
    float* feats = wsf + o;  o += (size_t)SEQT * LBL;       // 8192*512 f32
    float* Et    = wsf + o;  o += (size_t)LBL * LBL;        // 512*512 f32
    float* mT    = wsf + o;  o += LBL;
    float* gold  = wsf + o;  o += 16;                       // padded for alignment
    unsigned short* xbf = (unsigned short*)(wsf + o);  o += (size_t)SEQT * DIM / 2;
    unsigned short* Wt  = (unsigned short*)(wsf + o);  o += (size_t)DIM * LBL / 2;
    float* out = (float*)d_out;

    k_conv_x<<<dim3(SEQT * DIM / (8 * 256)), dim3(256), 0, stream>>>(x, xbf);
    k_conv_wt<<<dim3(DIM / 256, LBL), dim3(256), 0, stream>>>(W, Wt);

    dim3 gg(SEQT / 64, LBL / 64);
    k_gemm_feats<<<gg, dim3(256), 0, stream>>>(xbf, Wt, bias, feats);
    k_prep<<<dim3(LBL), dim3(256), 0, stream>>>(trans, Et, mT);
    k_gold<<<dim3(1), dim3(256), 0, stream>>>(feats, trans, tags, start_id, stop_id, gold);
    k_scan<<<dim3(1), dim3(512), 0, stream>>>(feats, Et, mT, trans, start_id, stop_id, gold, out);
}
